// CNN_GAT_66932770340988
// MI455X (gfx1250) — matrix operand order repaired and checked by reading, not compiled, not run on hardware
//
#include <hip/hip_runtime.h>

// ---------------------------------------------------------------------------
// Types / helpers
// ---------------------------------------------------------------------------
typedef __attribute__((ext_vector_type(16))) __bf16 v16bf;
typedef __attribute__((ext_vector_type(8)))  float  v8f;

__device__ __forceinline__ unsigned short f2bf(float f) {
  unsigned u = __float_as_uint(f);
  u += 0x7FFFu + ((u >> 16) & 1u);          // round-to-nearest-even
  return (unsigned short)(u >> 16);
}
__device__ __forceinline__ float bf2f(unsigned short h) {
  return __uint_as_float(((unsigned)h) << 16);
}
__device__ __forceinline__ v16bf frag_from(uint4 a, uint4 b) {
  union { v16bf v; uint4 q[2]; } u; u.q[0] = a; u.q[1] = b; return u.v;
}
// A-matrix 16x32 bf16 fragment: lane-half g; elems 0..7 = K g*8..g*8+7,
// elems 8..15 = K 16+g*8..16+g*8+7  (ISA 7.12.2)
__device__ __forceinline__ v16bf load_fragA(const unsigned short* row, int g) {
  return frag_from(*(const uint4*)(row + g * 8),
                   *(const uint4*)(row + 16 + g * 8));
}
// B-matrix 32x16 bf16 fragment (LDS stored n-major): lane-half g; elems 0..7 =
// K g*8..g*8+7, elems 8..15 = K 16+g*8..16+g*8+7 (same as A)
__device__ __forceinline__ v16bf load_fragB(const unsigned short* row, int g) {
  return frag_from(*(const uint4*)(row + g * 8),
                   *(const uint4*)(row + 16 + g * 8));
}

#define ROWS 16384   // 64*256 == 32*512

// ---------------------------------------------------------------------------
// Weight prep / embedding
// ---------------------------------------------------------------------------
__global__ void cvt_f32_bf16(const float* __restrict__ s,
                             unsigned short* __restrict__ d, int n) {
  int i = blockIdx.x * blockDim.x + threadIdx.x;
  if (i < n) d[i] = f2bf(s[i]);
}
// conv_w (co,ci,k)[256,256,5] -> B layout [k*256+ci][co] bf16
__global__ void cvt_conv_w(const float* __restrict__ s,
                           unsigned short* __restrict__ d) {
  int i = blockIdx.x * blockDim.x + threadIdx.x;   // over 1280*256
  int co = i & 255, rest = i >> 8, ci = rest & 255, k = rest >> 8;
  d[i] = f2bf(s[co * 1280 + ci * 5 + k]);
}
__global__ void embed_k(const int* __restrict__ ids,
                        const float* __restrict__ table,
                        unsigned short* __restrict__ eb) {
  int i = blockIdx.x * blockDim.x + threadIdx.x;   // over ROWS*256
  int r = i >> 8, c = i & 255;
  eb[i] = f2bf(table[(size_t)ids[r] * 256 + c]);
}

// ---------------------------------------------------------------------------
// WMMA GEMM: C[M,N] = A[M,K] * B[K,N], bf16 in, f32 accumulate.
// Block tile 256x64, K step 32, double-buffered LDS.
// 8 waves; each wave = 32 rows x 64 cols = 8 WMMAs per K-step.
// CONV: A addressed as on-the-fly im2col of emb (SAME pad, K=5, T=256).
// ---------------------------------------------------------------------------
template <bool CONV>
__device__ __forceinline__ void load_A_regs(const unsigned short* __restrict__ A,
                                            int m0, int t, int k0, int K,
                                            uint4 a[4]) {
  if (CONV) {
    int r = m0 + t, b2 = r >> 8, tp = r & 255;
    int kf = k0 >> 8, ci = k0 & 255;       // 32-chunk never crosses a k boundary
    int ts = tp + kf - 2;
    if (ts >= 0 && ts < 256) {
      const uint4* p = (const uint4*)(A + ((size_t)(b2 * 256 + ts)) * 256 + ci);
      a[0] = p[0]; a[1] = p[1]; a[2] = p[2]; a[3] = p[3];
    } else {
      uint4 z = make_uint4(0, 0, 0, 0);
      a[0] = z; a[1] = z; a[2] = z; a[3] = z;
    }
  } else {
    const uint4* p = (const uint4*)(A + (size_t)(m0 + t) * K + k0);
    a[0] = p[0]; a[1] = p[1]; a[2] = p[2]; a[3] = p[3];
  }
}

template <bool CONV, bool OUTBF>
__global__ __launch_bounds__(256) void gemm_wmma(
    const unsigned short* __restrict__ A, const unsigned short* __restrict__ Bw,
    float* __restrict__ Cf, unsigned short* __restrict__ Cb,
    int M, int N, int K) {
  __shared__ __align__(16) unsigned short As[2][256 * 32];
  __shared__ __align__(16) unsigned short Bs[2][64 * 32];
  const int t = threadIdx.x;
  const int n0 = blockIdx.x * 64, m0 = blockIdx.y * 256;
  const int lane = t & 31, w = t >> 5, g = lane >> 4, mr = lane & 15;
  const int brow = t >> 3, bseg = t & 7;         // B loader: 8 thr/row of 64

  v8f acc[2][4];
  v8f z = {0.f, 0.f, 0.f, 0.f, 0.f, 0.f, 0.f, 0.f};
#pragma unroll
  for (int h = 0; h < 2; h++)
#pragma unroll
    for (int i = 0; i < 4; i++) acc[h][i] = z;

  const int nk = K >> 5;
  uint4 ar[4], br;
  load_A_regs<CONV>(A, m0, t, 0, K, ar);
  br = *(const uint4*)(Bw + (size_t)brow * N + n0 + bseg * 8);
  {
    uint4* ad = (uint4*)(&As[0][t * 32]);
    ad[0] = ar[0]; ad[1] = ar[1]; ad[2] = ar[2]; ad[3] = ar[3];
    const unsigned short* be = (const unsigned short*)&br;
#pragma unroll
    for (int j = 0; j < 8; j++) Bs[0][(bseg * 8 + j) * 32 + brow] = be[j];
  }

  int cur = 0;
  for (int kt = 0; kt < nk; kt++) {
    __syncthreads();
    bool more = (kt + 1) < nk;
    if (more) {
      load_A_regs<CONV>(A, m0, t, (kt + 1) << 5, K, ar);
      br = *(const uint4*)(Bw + (size_t)(((kt + 1) << 5) + brow) * N + n0 + bseg * 8);
    }
    v16bf af0 = load_fragA(&As[cur][(w * 32 + mr) * 32], g);
    v16bf af1 = load_fragA(&As[cur][(w * 32 + 16 + mr) * 32], g);
#pragma unroll
    for (int nt = 0; nt < 4; nt++) {
      v16bf bfv = load_fragB(&Bs[cur][(nt * 16 + mr) * 32], g);
      acc[0][nt] = __builtin_amdgcn_wmma_f32_16x16x32_bf16(
          false, af0, false, bfv, (short)0, acc[0][nt], false, false);
      acc[1][nt] = __builtin_amdgcn_wmma_f32_16x16x32_bf16(
          false, af1, false, bfv, (short)0, acc[1][nt], false, false);
    }
    if (more) {
      uint4* ad = (uint4*)(&As[cur ^ 1][t * 32]);
      ad[0] = ar[0]; ad[1] = ar[1]; ad[2] = ar[2]; ad[3] = ar[3];
      const unsigned short* be = (const unsigned short*)&br;
#pragma unroll
      for (int j = 0; j < 8; j++) Bs[cur ^ 1][(bseg * 8 + j) * 32 + brow] = be[j];
    }
    cur ^= 1;
  }
#pragma unroll
  for (int h = 0; h < 2; h++)
#pragma unroll
    for (int nt = 0; nt < 4; nt++)
#pragma unroll
      for (int r8 = 0; r8 < 8; r8++) {
        int row = m0 + w * 32 + h * 16 + g * 8 + r8;
        int col = n0 + nt * 16 + mr;
        float v = acc[h][nt][r8];
        if (OUTBF) Cb[(size_t)row * N + col] = f2bf(v);
        else       Cf[(size_t)row * N + col] = v;
      }
}

// ---------------------------------------------------------------------------
// conv+bias+relu+mask + residual+bias, then LayerNorm over 256 cols
// ---------------------------------------------------------------------------
__global__ __launch_bounds__(256) void fuse_ln(
    const float* __restrict__ convo, const float* __restrict__ reso,
    const float* __restrict__ cb, const float* __restrict__ rb,
    const float* __restrict__ mask, const float* __restrict__ ng,
    const float* __restrict__ nb, float* __restrict__ xf,
    unsigned short* __restrict__ xb) {
  __shared__ float red[256];
  int r = blockIdx.x, c = threadIdx.x;
  size_t idx = (size_t)r * 256 + c;
  float v = convo[idx] + cb[c];
  v = v > 0.f ? v : 0.f;
  v *= mask[r];
  v += reso[idx] + rb[c];
  red[c] = v; __syncthreads();
  for (int s = 128; s > 0; s >>= 1) { if (c < s) red[c] += red[c + s]; __syncthreads(); }
  float mean = red[0] / 256.f; __syncthreads();
  float d = v - mean;
  red[c] = d * d; __syncthreads();
  for (int s = 128; s > 0; s >>= 1) { if (c < s) red[c] += red[c + s]; __syncthreads(); }
  float var = red[0] / 256.f;
  float xn = d * rsqrtf(var + 1e-5f) * ng[c] + nb[c];
  xf[idx] = xn; xb[idx] = f2bf(xn);
}

// ---------------------------------------------------------------------------
// GAT: per-(node,head) attention source/dest scores from bf16 h
// ---------------------------------------------------------------------------
__global__ void gat_esed(const unsigned short* __restrict__ h,
                         const float* __restrict__ asrc,
                         const float* __restrict__ adst,
                         float* __restrict__ es, float* __restrict__ ed) {
  int idx = blockIdx.x * blockDim.x + threadIdx.x;   // over B*N*H
  if (idx >= 32 * 512 * 8) return;
  int head = idx & 7, n = idx >> 3;
  const uint4* hp = (const uint4*)(h + (size_t)n * 256 + head * 32);
  float s1 = 0.f, s2 = 0.f;
#pragma unroll
  for (int q = 0; q < 4; q++) {
    uint4 hv = hp[q];
    const unsigned short* he = (const unsigned short*)&hv;
#pragma unroll
    for (int e = 0; e < 8; e++) {
      float f = bf2f(he[e]);
      int d = q * 8 + e;
      s1 += f * asrc[head * 32 + d];
      s2 += f * adst[head * 32 + d];
    }
  }
  es[idx] = s1; ed[idx] = s2;
}

// ---------------------------------------------------------------------------
// GAT pass A: row max + softmax denominator. Block = (b, 4 rows).
// ed slab [512][8] staged to LDS transposed [head][j]; 8 waves = 8 heads.
// ---------------------------------------------------------------------------
__global__ __launch_bounds__(256) void gat_ms(
    const float* __restrict__ adj, const float* __restrict__ es,
    const float* __restrict__ ed, float* __restrict__ mo, float* __restrict__ so) {
  __shared__ float eds[8 * 512];   // [head][j]
  __shared__ float ar[4 * 512];
  int b = blockIdx.y, i0 = blockIdx.x * 4, t = threadIdx.x;
  const float4* edg = (const float4*)(ed + (size_t)b * 4096);
#pragma unroll
  for (int q = 0; q < 4; q++) {
    float4 v = edg[t * 4 + q];
    int base = t * 16 + q * 4;
    eds[(((base + 0) & 7) << 9) + ((base + 0) >> 3)] = v.x;
    eds[(((base + 1) & 7) << 9) + ((base + 1) >> 3)] = v.y;
    eds[(((base + 2) & 7) << 9) + ((base + 2) >> 3)] = v.z;
    eds[(((base + 3) & 7) << 9) + ((base + 3) >> 3)] = v.w;
  }
  const float4* ag = (const float4*)(adj + ((size_t)(b * 512 + i0)) * 512);
#pragma unroll
  for (int q = 0; q < 2; q++) ((float4*)ar)[t * 2 + q] = ag[t * 2 + q];
  __syncthreads();

  int head = t >> 5, lane = t & 31;
  for (int il = 0; il < 4; il++) {
    float es_i = es[(size_t)(b * 512 + i0 + il) * 8 + head];
    float ev[16], okm[16];
    float mx = -3e38f;
#pragma unroll
    for (int q = 0; q < 16; q++) {
      int j = lane + q * 32;
      float e = es_i + eds[(head << 9) + j];
      e = e > 0.f ? e : 0.2f * e;
      bool ok = ar[(il << 9) + j] > 0.f;
      ev[q] = e; okm[q] = ok ? 1.f : 0.f;
      mx = fmaxf(mx, ok ? e : -3e38f);
    }
    for (int off = 16; off; off >>= 1) mx = fmaxf(mx, __shfl_xor(mx, off, 32));
    float sum = 0.f;
#pragma unroll
    for (int q = 0; q < 16; q++) sum += okm[q] * __expf(ev[q] - mx);
    for (int off = 16; off; off >>= 1) sum += __shfl_xor(sum, off, 32);
    if (lane == 0) {
      mo[(size_t)(b * 512 + i0 + il) * 8 + head] = mx;
      so[(size_t)(b * 512 + i0 + il) * 8 + head] = sum;
    }
  }
}

// ---------------------------------------------------------------------------
// GAT pass B: out[b,i,head,:] = sum_j P_ij * h[b,j,head,:] via WMMA.
// Block = (i-tile of 128, one head, one graph). j-step 64 => 4 WMMAs per
// barrier pair. ed staged in LDS; branchless P build; b128 P stores.
// ---------------------------------------------------------------------------
__global__ __launch_bounds__(256) void gat_pv(
    const float* __restrict__ adj, const unsigned short* __restrict__ h,
    const float* __restrict__ es, const float* __restrict__ ed,
    const float* __restrict__ mv, const float* __restrict__ sv,
    float* __restrict__ out) {
  __shared__ __align__(16) unsigned short Ps[128 * 64];
  __shared__ __align__(16) unsigned short Hs[32 * 64];
  __shared__ float eds[512];
  const int t = threadIdx.x;
  const int b = blockIdx.z, head = blockIdx.y, i0 = blockIdx.x * 128;
  const int hd0 = head * 32;
  const int lane = t & 31, w = t >> 5, g = lane >> 4, mr = lane & 15;
  const int il = t >> 1, jh = t & 1;               // P builder: 2 thr/row, 32 j each
  const int jb = t >> 2, cs = t & 3;               // H loader: 64 rows x 8 cols
  eds[t]       = ed[((size_t)b * 512 + t) * 8 + head];
  eds[t + 256] = ed[((size_t)b * 512 + t + 256) * 8 + head];

  const size_t gi = (size_t)b * 512 + i0 + il;
  const float es_i  = es[gi * 8 + head];
  const float m_i   = mv[gi * 8 + head];
  const float inv_s = 1.0f / sv[gi * 8 + head];

  v8f acc[2];
  v8f z = {0.f, 0.f, 0.f, 0.f, 0.f, 0.f, 0.f, 0.f};
  acc[0] = z; acc[1] = z;
  __syncthreads();

  for (int j0 = 0; j0 < 512; j0 += 64) {
    const float4* arow4 = (const float4*)(adj + gi * 512 + j0 + jh * 32);
    if (j0 + 64 < 512) __builtin_prefetch(adj + gi * 512 + j0 + 64, 0, 1);
    unsigned short pb[32];
#pragma unroll
    for (int q = 0; q < 8; q++) {
      float4 a4 = arow4[q];
      const float* ac = (const float*)&a4;
#pragma unroll
      for (int c = 0; c < 4; c++) {
        int j = j0 + jh * 32 + q * 4 + c;
        float e = es_i + eds[j];
        e = e > 0.f ? e : 0.2f * e;
        float p = __expf(e - m_i) * inv_s;
        pb[q * 4 + c] = f2bf(ac[c] > 0.f ? p : 0.f);
      }
    }
    {
      const uint4* pv = (const uint4*)pb;
      uint4* dst = (uint4*)(Ps + il * 64 + jh * 32);
      dst[0] = pv[0]; dst[1] = pv[1]; dst[2] = pv[2]; dst[3] = pv[3];
    }
    uint4 hv = *(const uint4*)(h + ((size_t)(b * 512 + j0 + jb)) * 256 + hd0 + cs * 8);
    const unsigned short* he = (const unsigned short*)&hv;
#pragma unroll
    for (int cc = 0; cc < 8; cc++) Hs[(cs * 8 + cc) * 64 + jb] = he[cc];
    __syncthreads();

#pragma unroll
    for (int kc = 0; kc < 2; kc++) {
      v16bf af = load_fragA(Ps + (w * 16 + mr) * 64 + kc * 32, g);
#pragma unroll
      for (int nt = 0; nt < 2; nt++) {
        v16bf bfv = load_fragB(Hs + (nt * 16 + mr) * 64 + kc * 32, g);
        acc[nt] = __builtin_amdgcn_wmma_f32_16x16x32_bf16(
            false, af, false, bfv, (short)0, acc[nt], false, false);
      }
    }
    __syncthreads();
  }
#pragma unroll
  for (int nt = 0; nt < 2; nt++)
#pragma unroll
    for (int r8 = 0; r8 < 8; r8++) {
      size_t row = (size_t)b * 512 + i0 + w * 16 + g * 8 + r8;
      out[row * 256 + hd0 + nt * 16 + mr] = acc[nt][r8];
    }
}

// x = relu(inter + b_i) + relu(outer + b_o); write f32 + bf16
__global__ void combine_k(const float* __restrict__ tA, const float* __restrict__ tB,
                          const float* __restrict__ bi, const float* __restrict__ bo,
                          float* __restrict__ xf, unsigned short* __restrict__ xb) {
  size_t idx = (size_t)blockIdx.x * blockDim.x + threadIdx.x;  // over B*N*256
  int c = (int)(idx & 255);
  float a = tA[idx] + bi[c]; a = a > 0.f ? a : 0.f;
  float o = tB[idx] + bo[c]; o = o > 0.f ? o : 0.f;
  float x = a + o;
  xf[idx] = x; xb[idx] = f2bf(x);
}

// two-stage max readout over nodes
__global__ void pool_part(const float* __restrict__ xf, float* __restrict__ part) {
  int b = blockIdx.x, gq = blockIdx.y, c = threadIdx.x;
  float mx = -3e38f;
  for (int n = gq * 32; n < gq * 32 + 32; n++)
    mx = fmaxf(mx, xf[((size_t)(b * 512 + n)) * 256 + c]);
  part[((size_t)(b * 16 + gq)) * 256 + c] = mx;
}
__global__ void pool_fin(const float* __restrict__ part, float* __restrict__ pooled,
                         int off) {
  int b = blockIdx.x, c = threadIdx.x;
  float mx = -3e38f;
  for (int q = 0; q < 16; q++)
    mx = fmaxf(mx, part[((size_t)(b * 16 + q)) * 256 + c]);
  pooled[b * 512 + off + c] = mx;
}

// LayerNorm(512) -> ELU MLP(512->256) -> (256->2) -> log_softmax
__global__ __launch_bounds__(256) void head_k(
    const float* __restrict__ pooled, const float* __restrict__ cg,
    const float* __restrict__ cbv, const float* __restrict__ w1,
    const float* __restrict__ b1, const float* __restrict__ w2,
    const float* __restrict__ b2, float* __restrict__ out) {
  __shared__ float pl[512];
  __shared__ float red[256];
  __shared__ float h1[256];
  __shared__ float lg[2];
  int b = blockIdx.x, t = threadIdx.x;
  pl[t] = pooled[b * 512 + t]; pl[t + 256] = pooled[b * 512 + t + 256];
  __syncthreads();
  red[t] = pl[t] + pl[t + 256]; __syncthreads();
  for (int s = 128; s > 0; s >>= 1) { if (t < s) red[t] += red[t + s]; __syncthreads(); }
  float mean = red[0] / 512.f; __syncthreads();
  float d0 = pl[t] - mean, d1 = pl[t + 256] - mean;
  red[t] = d0 * d0 + d1 * d1; __syncthreads();
  for (int s = 128; s > 0; s >>= 1) { if (t < s) red[t] += red[t + s]; __syncthreads(); }
  float rs = rsqrtf(red[0] / 512.f + 1e-5f); __syncthreads();
  pl[t]       = d0 * rs * cg[t]       + cbv[t];
  pl[t + 256] = d1 * rs * cg[t + 256] + cbv[t + 256];
  __syncthreads();
  float a = b1[t];
  for (int j = 0; j < 512; j++) a += pl[j] * w1[j * 256 + t];
  h1[t] = a > 0.f ? a : (__expf(a) - 1.f);
  __syncthreads();
  if (t < 2) {
    float l = b2[t];
    for (int c = 0; c < 256; c++) l += h1[c] * w2[c * 2 + t];
    lg[t] = l;
  }
  __syncthreads();
  if (t < 2) {
    float m = fmaxf(lg[0], lg[1]);
    float lse = m + __logf(__expf(lg[0] - m) + __expf(lg[1] - m));
    out[b * 2 + t] = lg[t] - lse;
  }
}

// ---------------------------------------------------------------------------
// Launcher
// ---------------------------------------------------------------------------
extern "C" void kernel_launch(void* const* d_in, const int* in_sizes, int n_in,
                              void* d_out, int out_size, void* d_ws, size_t ws_size,
                              hipStream_t stream) {
  const int*   ids   = (const int*)d_in[0];
  const float* mask  = (const float*)d_in[1];
  const float* adjI  = (const float*)d_in[2];
  const float* adjO  = (const float*)d_in[3];
  const float* embT  = (const float*)d_in[4];
  const float* convw = (const float*)d_in[5];
  const float* convb = (const float*)d_in[6];
  const float* resw  = (const float*)d_in[7];
  const float* resb  = (const float*)d_in[8];
  const float* ng    = (const float*)d_in[9];
  const float* nbv   = (const float*)d_in[10];
  const float* cng   = (const float*)d_in[11];
  const float* cnb   = (const float*)d_in[12];
  const float* pw1   = (const float*)d_in[13];
  const float* pb1   = (const float*)d_in[14];
  const float* pw2   = (const float*)d_in[15];
  const float* pb2   = (const float*)d_in[16];
  const float* gW[4];  const float* gAs[4]; const float* gAd[4]; const float* gB[4];
  for (int s = 0; s < 4; s++) {
    gW[s]  = (const float*)d_in[17 + 4 * s + 0];
    gAs[s] = (const float*)d_in[17 + 4 * s + 1];
    gAd[s] = (const float*)d_in[17 + 4 * s + 2];
    gB[s]  = (const float*)d_in[17 + 4 * s + 3];
  }

  char* w = (char*)d_ws;
  size_t off = 0;
  auto alloc = [&](size_t bytes) { char* p = w + off; off += (bytes + 255) & ~(size_t)255; return p; };
  unsigned short* emb_bf = (unsigned short*)alloc((size_t)ROWS * 256 * 2);
  unsigned short* wBconv = (unsigned short*)alloc((size_t)1280 * 256 * 2);
  unsigned short* wBres  = (unsigned short*)alloc((size_t)256 * 256 * 2);
  unsigned short* wBgat[4];
  for (int s = 0; s < 4; s++) wBgat[s] = (unsigned short*)alloc((size_t)256 * 256 * 2);
  float*          x_f    = (float*)alloc((size_t)ROWS * 256 * 4);
  unsigned short* x_bf   = (unsigned short*)alloc((size_t)ROWS * 256 * 2);
  float*          tmpA   = (float*)alloc((size_t)ROWS * 256 * 4);
  float*          tmpB   = (float*)alloc((size_t)ROWS * 256 * 4);
  unsigned short* h_i    = (unsigned short*)alloc((size_t)ROWS * 256 * 2);
  unsigned short* h_o    = (unsigned short*)alloc((size_t)ROWS * 256 * 2);
  float* es_i = (float*)alloc((size_t)ROWS * 8 * 4);
  float* ed_i = (float*)alloc((size_t)ROWS * 8 * 4);
  float* es_o = (float*)alloc((size_t)ROWS * 8 * 4);
  float* ed_o = (float*)alloc((size_t)ROWS * 8 * 4);
  float* m_i  = (float*)alloc((size_t)ROWS * 8 * 4);
  float* s_i  = (float*)alloc((size_t)ROWS * 8 * 4);
  float* m_o  = (float*)alloc((size_t)ROWS * 8 * 4);
  float* s_o  = (float*)alloc((size_t)ROWS * 8 * 4);
  float* pooled = (float*)alloc((size_t)32 * 512 * 4);
  float* part   = (float*)alloc((size_t)32 * 16 * 256 * 4);

  // weights -> bf16
  cvt_conv_w<<<1280 * 256 / 256, 256, 0, stream>>>(convw, wBconv);
  cvt_f32_bf16<<<256, 256, 0, stream>>>(resw, wBres, 65536);
  for (int s = 0; s < 4; s++)
    cvt_f32_bf16<<<256, 256, 0, stream>>>(gW[s], wBgat[s], 65536);

  // embedding gather
  embed_k<<<ROWS, 256, 0, stream>>>(ids, embT, emb_bf);

  // conv (implicit-GEMM) and residual projection
  gemm_wmma<true, false><<<dim3(4, 64), 256, 0, stream>>>(
      emb_bf, wBconv, tmpA, nullptr, ROWS, 256, 1280);
  gemm_wmma<false, false><<<dim3(4, 64), 256, 0, stream>>>(
      emb_bf, wBres, tmpB, nullptr, ROWS, 256, 256);
  fuse_ln<<<ROWS, 256, 0, stream>>>(tmpA, tmpB, convb, resb, mask, ng, nbv, x_f, x_bf);

  // two GAT layers, each inter+outer
  for (int li = 0; li < 2; li++) {
    int si = li * 2 + 0, so = li * 2 + 1;
    gemm_wmma<false, true><<<dim3(4, 64), 256, 0, stream>>>(
        x_bf, wBgat[si], nullptr, h_i, ROWS, 256, 256);
    gat_esed<<<ROWS * 8 / 256, 256, 0, stream>>>(h_i, gAs[si], gAd[si], es_i, ed_i);
    gat_ms<<<dim3(128, 32), 256, 0, stream>>>(adjI, es_i, ed_i, m_i, s_i);
    gat_pv<<<dim3(4, 8, 32), 256, 0, stream>>>(adjI, h_i, es_i, ed_i, m_i, s_i, tmpA);

    gemm_wmma<false, true><<<dim3(4, 64), 256, 0, stream>>>(
        x_bf, wBgat[so], nullptr, h_o, ROWS, 256, 256);
    gat_esed<<<ROWS * 8 / 256, 256, 0, stream>>>(h_o, gAs[so], gAd[so], es_o, ed_o);
    gat_ms<<<dim3(128, 32), 256, 0, stream>>>(adjO, es_o, ed_o, m_o, s_o);
    gat_pv<<<dim3(4, 8, 32), 256, 0, stream>>>(adjO, h_o, es_o, ed_o, m_o, s_o, tmpB);

    combine_k<<<ROWS, 256, 0, stream>>>(tmpA, tmpB, gB[si], gB[so], x_f, x_bf);
    pool_part<<<dim3(32, 16), 256, 0, stream>>>(x_f, part);
    pool_fin<<<32, 256, 0, stream>>>(part, pooled, li * 256);
  }

  head_k<<<32, 256, 0, stream>>>(pooled, cng, cnb, pw1, pb1, pw2, pb2, (float*)d_out);
}
